// SelfAttention_40630390620216
// MI455X (gfx1250) — matrix operand order repaired and checked by reading, not compiled, not run on hardware
//
#include <hip/hip_runtime.h>

// ---- problem constants (match reference) ----
#define D_MODEL 1024
#define NHEAD   16
#define DHEAD   64
#define SEQ     2048
#define BATCH   2
#define ROWS    (BATCH * SEQ)   // 4096
#define LN_EPS  1e-5f

typedef _Float16 f16;
typedef _Float16 v16h __attribute__((ext_vector_type(16)));
typedef float    v8f  __attribute__((ext_vector_type(8)));
typedef unsigned int u32;
typedef u32 u32x4 __attribute__((ext_vector_type(4)));
typedef u32 u32x2 __attribute__((ext_vector_type(2)));

union FragH { v16h h; u32x4 q[2]; };
union Pack4 { f16 h[4]; u32x2 u; };

// ------------------------------------------------------------------
// WMMA wrapper: D = A(16x32 f16) * B(32x16 f16) + C(16x16 f32)
// ------------------------------------------------------------------
__device__ __forceinline__ v8f wmma_f16(v16h a, v16h b, v8f c) {
  return __builtin_amdgcn_wmma_f32_16x16x32_f16(
      /*neg_a=*/false, a, /*neg_b=*/false, b,
      /*c_mod=*/(short)0, c, /*reuse_a=*/false, /*reuse_b=*/false);
}

// A-matrix fragment (16x32, MxK, f16). ISA layout: lane<16 holds row M=lane
// with K = {0..7, 16..23}; lane>=16 holds same rows with K = {8..15, 24..31}.
__device__ __forceinline__ v16h load_fragA(const f16* base, int ld,
                                           int m_base, int k_base, int lane) {
  const int r  = m_base + (lane & 15);
  const int kb = k_base + ((lane >> 4) << 3);         // 0 or 8
  const f16* p = base + (size_t)r * ld + kb;
  FragH f;
  f.q[0] = *(const u32x4*)(p);                        // K = kb .. kb+7
  f.q[1] = *(const u32x4*)(p + 16);                   // K = kb+16 .. kb+23
  return f.h;
}

// B-matrix fragment (32x16, KxN, f16); we need B[k][n] = SRC[n][k], SRC rows
// contiguous in k.  Lane<16: N=lane, K={0..7,16..23}; lane>=16: N=lane-16,
// K={8..15,24..31}.
__device__ __forceinline__ v16h load_fragB(const f16* base, int ld,
                                           int n_base, int k_base, int lane) {
  const int n  = n_base + (lane & 15);
  const int kb = k_base + ((lane >> 4) << 3);         // 0 or 8
  const f16* p = base + (size_t)n * ld + kb;
  FragH f;
  f.q[0] = *(const u32x4*)(p);
  f.q[1] = *(const u32x4*)(p + 16);
  return f.h;
}

// ------------------------------------------------------------------
// CDNA5 async global->LDS copy (ASYNCcnt tracked) + waits
// ------------------------------------------------------------------
__device__ __forceinline__ u32 lds_addr32(const void* p) {
  // flat LDS pointers carry the in-LDS byte offset in the low 32 bits
  return (u32)(size_t)p;
}
__device__ __forceinline__ void async_copy_b128(u32 lds_off, const void* gptr) {
  asm volatile("global_load_async_to_lds_b128 %0, %1, off"
               :: "v"(lds_off), "v"((unsigned long long)(size_t)gptr)
               : "memory");
}
__device__ __forceinline__ void wait_async0() {
  asm volatile("s_wait_asynccnt 0x0" ::: "memory");
}
__device__ __forceinline__ void wait_ds0() {
  asm volatile("s_wait_dscnt 0x0" ::: "memory");
}

// ------------------------------------------------------------------
// Kernel 0: f32 -> f16 weight cast (vectorized x4)
// ------------------------------------------------------------------
__global__ __launch_bounds__(256) void k_cast_f16(const float* __restrict__ src,
                                                  f16* __restrict__ dst, int n4) {
  int i = (blockIdx.x * 256 + threadIdx.x);
  if (i >= n4) return;
  const float4 v = *(const float4*)(src + (size_t)i * 4);
  Pack4 o;
  o.h[0] = (f16)v.x; o.h[1] = (f16)v.y; o.h[2] = (f16)v.z; o.h[3] = (f16)v.w;
  *(u32x2*)(dst + (size_t)i * 4) = o.u;
}

// ------------------------------------------------------------------
// Kernel 1: LayerNorm, one wave32 per row, h -> f16
// ------------------------------------------------------------------
__global__ __launch_bounds__(128) void k_layernorm(const float* __restrict__ x,
                                                   const float* __restrict__ gamma,
                                                   const float* __restrict__ beta,
                                                   f16* __restrict__ h) {
  const int lane = threadIdx.x & 31;
  const int row  = blockIdx.x * 4 + (threadIdx.x >> 5);
  const float* xr = x + (size_t)row * D_MODEL;

  float4 v[8];
  float s = 0.f, sq = 0.f;
#pragma unroll
  for (int i = 0; i < 8; ++i) {
    v[i] = *(const float4*)(xr + (i * 32 + lane) * 4);
    const float* pv = &v[i].x;
#pragma unroll
    for (int j = 0; j < 4; ++j) { s += pv[j]; sq += pv[j] * pv[j]; }
  }
#pragma unroll
  for (int m = 16; m; m >>= 1) {
    s  += __shfl_xor(s,  m, 32);
    sq += __shfl_xor(sq, m, 32);
  }
  const float mu  = s * (1.0f / D_MODEL);
  const float var = sq * (1.0f / D_MODEL) - mu * mu;
  const float rs  = rsqrtf(var + LN_EPS);

  f16* hr = h + (size_t)row * D_MODEL;
#pragma unroll
  for (int i = 0; i < 8; ++i) {
    const int c0 = (i * 32 + lane) * 4;
    const float* pv = &v[i].x;
    Pack4 o;
#pragma unroll
    for (int j = 0; j < 4; ++j)
      o.h[j] = (f16)((pv[j] - mu) * rs * gamma[c0 + j] + beta[c0 + j]);
    *(u32x2*)(hr + c0) = o.u;
  }
}

// ------------------------------------------------------------------
// Kernel 2: fused QKV GEMM  y = h @ W.T + b   (blockIdx.z selects Q/K/V)
// Double-buffered async global->LDS staging of A(128x32) and B(128x32) tiles.
// Block: 256 thr = 8 waves (4 along M x 2 along N); wave tile 32x64.
// ------------------------------------------------------------------
__global__ __launch_bounds__(256) void k_gemm_qkv(
    const f16* __restrict__ h,
    const f16* __restrict__ wq, const f16* __restrict__ wk, const f16* __restrict__ wv,
    const float* __restrict__ bq, const float* __restrict__ bk, const float* __restrict__ bv,
    f16* __restrict__ qo, f16* __restrict__ ko, f16* __restrict__ vto) {
  __shared__ f16 Abuf[2][128 * 32];
  __shared__ f16 Bbuf[2][128 * 32];

  const int tid  = threadIdx.x;
  const int lane = tid & 31;
  const int wave = tid >> 5;
  const int mloc = (wave & 3) * 32;
  const int nloc = (wave >> 2) * 64;
  const int m_blk = blockIdx.x * 128;
  const int n_blk = blockIdx.y * 128;
  const int z  = blockIdx.z;
  const f16*   w    = (z == 0) ? wq : (z == 1) ? wk : wv;
  const float* bias = (z == 0) ? bq : (z == 1) ? bk : bv;

  v8f acc[2][4];
#pragma unroll
  for (int mt = 0; mt < 2; ++mt)
#pragma unroll
    for (int nt = 0; nt < 4; ++nt) acc[mt][nt] = (v8f){};

  // issue async copies for one 128x32 A tile + 128x32 B tile (8KB each):
  // 512 16B chunks per tile, 2 per thread per tile.
  auto issue = [&](int buf, int k) {
    const u32 aoff = lds_addr32(&Abuf[buf][0]);
    const u32 boff = lds_addr32(&Bbuf[buf][0]);
#pragma unroll
    for (int t = 0; t < 2; ++t) {
      const int c   = tid + t * 256;      // 0..511
      const int row = c >> 2;             // 0..127
      const int ko  = (c & 3) * 8;        // halves within row
      async_copy_b128(aoff + row * 64 + ko * 2,
                      h + (size_t)(m_blk + row) * D_MODEL + k + ko);
      async_copy_b128(boff + row * 64 + ko * 2,
                      w + (size_t)(n_blk + row) * D_MODEL + k + ko);
    }
  };

  issue(0, 0);
  for (int kc = 0; kc < D_MODEL / 32; ++kc) {
    wait_async0();            // my copies for tile kc are in LDS
    __syncthreads();          // everyone's copies visible; prev buffer free
    if (kc + 1 < D_MODEL / 32) issue((kc + 1) & 1, (kc + 1) * 32);

    const f16* As = &Abuf[kc & 1][0];
    const f16* Bs = &Bbuf[kc & 1][0];
    const v16h a0 = load_fragA(As, 32, mloc,      0, lane);
    const v16h a1 = load_fragA(As, 32, mloc + 16, 0, lane);
    v16h b[4];
#pragma unroll
    for (int nt = 0; nt < 4; ++nt) b[nt] = load_fragB(Bs, 32, nloc + nt * 16, 0, lane);
#pragma unroll
    for (int nt = 0; nt < 4; ++nt) {
      acc[0][nt] = wmma_f16(a0, b[nt], acc[0][nt]);
      acc[1][nt] = wmma_f16(a1, b[nt], acc[1][nt]);
    }
  }

  const int g8 = (lane >> 4) << 3;
  const int cl = lane & 15;
#pragma unroll
  for (int nt = 0; nt < 4; ++nt) {
    const int col  = n_blk + nloc + nt * 16 + cl;
    const float bi = bias[col];
    const int head = col >> 6, d = col & 63;
#pragma unroll
    for (int mt = 0; mt < 2; ++mt)
#pragma unroll
      for (int j = 0; j < 8; ++j) {
        const int row = m_blk + mloc + mt * 16 + j + g8;
        const int bb = row >> 11, t = row & (SEQ - 1);
        const float val = acc[mt][nt][j] + bi;
        const size_t bh = (size_t)(bb * NHEAD + head);
        if (z == 0)      qo[(bh * SEQ + t) * DHEAD + d] = (f16)(val * 0.125f);
        else if (z == 1) ko[(bh * SEQ + t) * DHEAD + d] = (f16)val;
        else             vto[(bh * DHEAD + d) * SEQ + t] = (f16)val;
      }
  }
}

// ------------------------------------------------------------------
// Kernel 3: flash attention.  One wave per 16-query tile; the 4 waves of a
// block share (batch, head), so K/V s-block tiles are staged once per block
// into LDS with double-buffered async copies.  SiLU fused into the store.
// ------------------------------------------------------------------
__global__ __launch_bounds__(128) void k_attention(
    const f16* __restrict__ q, const f16* __restrict__ k,
    const f16* __restrict__ vt, f16* __restrict__ att) {
  __shared__ f16 Kbuf[2][32 * 64];   // [s'][d]   4KB each
  __shared__ f16 Vbuf[2][64 * 32];   // [d][s']   4KB each
  __shared__ f16 Pbuf[4][16 * 32];   // per-wave P re-swizzle

  const int tid  = threadIdx.x;
  const int lane = tid & 31;
  const int wid  = tid >> 5;
  const int gw   = blockIdx.x * 4 + wid;
  const int qt   = gw & 127;
  const int head = (gw >> 7) & (NHEAD - 1);
  const int b    = gw >> 11;
  const size_t bh = (size_t)(b * NHEAD + head);
  const f16* qb = q  + bh * SEQ * DHEAD;
  const f16* kb = k  + bh * SEQ * DHEAD;
  const f16* vb = vt + bh * DHEAD * SEQ;

  const v16h qa0 = load_fragA(qb, DHEAD, qt * 16, 0,  lane);
  const v16h qa1 = load_fragA(qb, DHEAD, qt * 16, 32, lane);

  v8f ov[4];
#pragma unroll
  for (int nt = 0; nt < 4; ++nt) ov[nt] = (v8f){};
  float mrow[8], lrow[8];
#pragma unroll
  for (int j = 0; j < 8; ++j) { mrow[j] = -1e30f; lrow[j] = 0.f; }

  f16* pl = Pbuf[wid];
  const int g8 = (lane >> 4) << 3;
  const int cl = lane & 15;

  // stage K (32x64 halves) and V^T (64x32 halves) tiles: 256 16B chunks each,
  // 2 per thread per tile.
  auto issue = [&](int buf, int s) {
    const u32 koff = lds_addr32(&Kbuf[buf][0]);
    const u32 voff = lds_addr32(&Vbuf[buf][0]);
#pragma unroll
    for (int t = 0; t < 2; ++t) {
      const int c = tid + t * 128;        // 0..255
      const int kr = c >> 3, kk = (c & 7) * 8;   // K: row=s', 8 chunks/row
      async_copy_b128(koff + kr * 128 + kk * 2,
                      kb + (size_t)(s + kr) * DHEAD + kk);
      const int vr = c >> 2, vk = (c & 3) * 8;   // V: row=d, 4 chunks/row
      async_copy_b128(voff + vr * 64 + vk * 2,
                      vb + (size_t)vr * SEQ + s + vk);
    }
  };

  issue(0, 0);
  for (int sb = 0; sb < SEQ / 32; ++sb) {
    const int s = sb * 32;
    wait_async0();
    __syncthreads();
    if (sb + 1 < SEQ / 32) issue((sb + 1) & 1, s + 32);

    const f16* Ks = &Kbuf[sb & 1][0];
    const f16* Vs = &Vbuf[sb & 1][0];

    // scores 16x32: two 16x16 N-tiles, K=dh=64 in two chunks
    const v16h kb0a = load_fragB(Ks, 64, 0,  0,  lane);
    const v16h kb0b = load_fragB(Ks, 64, 0,  32, lane);
    const v16h kb1a = load_fragB(Ks, 64, 16, 0,  lane);
    const v16h kb1b = load_fragB(Ks, 64, 16, 32, lane);
    v8f sc0 = wmma_f16(qa0, kb0a, (v8f){});
    sc0     = wmma_f16(qa1, kb0b, sc0);
    v8f sc1 = wmma_f16(qa0, kb1a, (v8f){});
    sc1     = wmma_f16(qa1, kb1b, sc1);

    // online softmax (per-row, segmented 16-lane reductions)
    float mnew[8], alpha[8];
#pragma unroll
    for (int j = 0; j < 8; ++j) {
      float mx = fmaxf(sc0[j], sc1[j]);
#pragma unroll
      for (int msk = 8; msk; msk >>= 1) mx = fmaxf(mx, __shfl_xor(mx, msk, 32));
      mnew[j]  = fmaxf(mrow[j], mx);
      alpha[j] = __expf(mrow[j] - mnew[j]);
      mrow[j]  = mnew[j];
    }
#pragma unroll
    for (int j = 0; j < 8; ++j) {
      const float p0 = __expf(sc0[j] - mnew[j]);
      const float p1 = __expf(sc1[j] - mnew[j]);
      float rs = p0 + p1;
#pragma unroll
      for (int msk = 8; msk; msk >>= 1) rs += __shfl_xor(rs, msk, 32);
      lrow[j] = lrow[j] * alpha[j] + rs;
      const int mr = j + g8;
      pl[mr * 32 + cl]      = (f16)p0;
      pl[mr * 32 + 16 + cl] = (f16)p1;
#pragma unroll
      for (int nt = 0; nt < 4; ++nt) ov[nt][j] *= alpha[j];
    }
    wait_ds0();   // per-wave LDS stores visible to same-wave reads

    // re-read P in A-layout, accumulate P @ V (V pre-transposed (d,t))
    const f16* pp = pl + cl * 32 + g8;
    FragH pf;
    pf.q[0] = *(const u32x4*)(pp);
    pf.q[1] = *(const u32x4*)(pp + 16);
    v16h vf[4];
#pragma unroll
    for (int nt = 0; nt < 4; ++nt) vf[nt] = load_fragB(Vs, 32, nt * 16, 0, lane);
#pragma unroll
    for (int nt = 0; nt < 4; ++nt) ov[nt] = wmma_f16(pf.h, vf[nt], ov[nt]);
  }

  // normalize, SiLU, store f16 activation for final projection
  float inv[8];
#pragma unroll
  for (int j = 0; j < 8; ++j) inv[j] = 1.0f / lrow[j];
#pragma unroll
  for (int nt = 0; nt < 4; ++nt) {
    const int col = head * 64 + nt * 16 + cl;
#pragma unroll
    for (int j = 0; j < 8; ++j) {
      const int trow = qt * 16 + j + g8;
      const float o  = ov[nt][j] * inv[j];
      const float sl = o / (1.0f + __expf(-o));   // SiLU
      att[((size_t)(b * SEQ + trow)) * D_MODEL + col] = (f16)sl;
    }
  }
}

// ------------------------------------------------------------------
// Kernel 4: output projection with residual: y = x + a @ Wo.T + bo (f32 out)
// Same async double-buffered LDS pipeline as the QKV GEMM.
// ------------------------------------------------------------------
__global__ __launch_bounds__(256) void k_gemm_out(
    const f16* __restrict__ a, const f16* __restrict__ wo,
    const float* __restrict__ bo, const float* __restrict__ x,
    float* __restrict__ y) {
  __shared__ f16 Abuf[2][128 * 32];
  __shared__ f16 Bbuf[2][128 * 32];

  const int tid  = threadIdx.x;
  const int lane = tid & 31;
  const int wave = tid >> 5;
  const int mloc = (wave & 3) * 32;
  const int nloc = (wave >> 2) * 64;
  const int m_blk = blockIdx.x * 128;
  const int n_blk = blockIdx.y * 128;

  v8f acc[2][4];
#pragma unroll
  for (int mt = 0; mt < 2; ++mt)
#pragma unroll
    for (int nt = 0; nt < 4; ++nt) acc[mt][nt] = (v8f){};

  auto issue = [&](int buf, int k) {
    const u32 aoff = lds_addr32(&Abuf[buf][0]);
    const u32 boff = lds_addr32(&Bbuf[buf][0]);
#pragma unroll
    for (int t = 0; t < 2; ++t) {
      const int c   = tid + t * 256;
      const int row = c >> 2;
      const int ko  = (c & 3) * 8;
      async_copy_b128(aoff + row * 64 + ko * 2,
                      a  + (size_t)(m_blk + row) * D_MODEL + k + ko);
      async_copy_b128(boff + row * 64 + ko * 2,
                      wo + (size_t)(n_blk + row) * D_MODEL + k + ko);
    }
  };

  issue(0, 0);
  for (int kc = 0; kc < D_MODEL / 32; ++kc) {
    wait_async0();
    __syncthreads();
    if (kc + 1 < D_MODEL / 32) issue((kc + 1) & 1, (kc + 1) * 32);

    const f16* As = &Abuf[kc & 1][0];
    const f16* Bs = &Bbuf[kc & 1][0];
    const v16h a0 = load_fragA(As, 32, mloc,      0, lane);
    const v16h a1 = load_fragA(As, 32, mloc + 16, 0, lane);
    v16h bfr[4];
#pragma unroll
    for (int nt = 0; nt < 4; ++nt) bfr[nt] = load_fragB(Bs, 32, nloc + nt * 16, 0, lane);
#pragma unroll
    for (int nt = 0; nt < 4; ++nt) {
      acc[0][nt] = wmma_f16(a0, bfr[nt], acc[0][nt]);
      acc[1][nt] = wmma_f16(a1, bfr[nt], acc[1][nt]);
    }
  }

  const int g8 = (lane >> 4) << 3;
  const int cl = lane & 15;
#pragma unroll
  for (int nt = 0; nt < 4; ++nt) {
    const int col  = n_blk + nloc + nt * 16 + cl;
    const float bi = bo[col];
#pragma unroll
    for (int mt = 0; mt < 2; ++mt)
#pragma unroll
      for (int j = 0; j < 8; ++j) {
        const int row = m_blk + mloc + mt * 16 + j + g8;
        const size_t idx = (size_t)row * D_MODEL + col;
        y[idx] = x[idx] + acc[mt][nt][j] + bi;
      }
  }
}

// ------------------------------------------------------------------
// Host launcher
// ------------------------------------------------------------------
extern "C" void kernel_launch(void* const* d_in, const int* in_sizes, int n_in,
                              void* d_out, int out_size, void* d_ws, size_t ws_size,
                              hipStream_t stream) {
  const float* x     = (const float*)d_in[0];
  const float* gamma = (const float*)d_in[1];
  const float* beta  = (const float*)d_in[2];
  const float* Wq    = (const float*)d_in[3];
  const float* bq    = (const float*)d_in[4];
  const float* Wk    = (const float*)d_in[5];
  const float* bk    = (const float*)d_in[6];
  const float* Wv    = (const float*)d_in[7];
  const float* bv    = (const float*)d_in[8];
  const float* Wo    = (const float*)d_in[9];
  const float* bo    = (const float*)d_in[10];
  float* y = (float*)d_out;

  // workspace carve-up (~48 MB total)
  char* ws = (char*)d_ws;
  const size_t szAct = (size_t)ROWS * D_MODEL * sizeof(f16);        // 8 MB
  const size_t szW   = (size_t)D_MODEL * D_MODEL * sizeof(f16);     // 2 MB
  f16* hf  = (f16*)ws;             ws += szAct;
  f16* wqh = (f16*)ws;             ws += szW;
  f16* wkh = (f16*)ws;             ws += szW;
  f16* wvh = (f16*)ws;             ws += szW;
  f16* woh = (f16*)ws;             ws += szW;
  f16* qf  = (f16*)ws;             ws += szAct;   // (b,h,t,d)
  f16* kf  = (f16*)ws;             ws += szAct;   // (b,h,t,d)
  f16* vtf = (f16*)ws;             ws += szAct;   // (b,h,d,t)
  f16* atf = (f16*)ws;             ws += szAct;   // silu(attn out), (row, D)

  const int n4 = (D_MODEL * D_MODEL) / 4;
  const int cb = (n4 + 255) / 256;
  k_cast_f16<<<cb, 256, 0, stream>>>(Wq, wqh, n4);
  k_cast_f16<<<cb, 256, 0, stream>>>(Wk, wkh, n4);
  k_cast_f16<<<cb, 256, 0, stream>>>(Wv, wvh, n4);
  k_cast_f16<<<cb, 256, 0, stream>>>(Wo, woh, n4);

  k_layernorm<<<ROWS / 4, 128, 0, stream>>>(x, gamma, beta, hf);

  dim3 gq(ROWS / 128, D_MODEL / 128, 3);            // (32, 8, 3)
  k_gemm_qkv<<<gq, 256, 0, stream>>>(hf, wqh, wkh, wvh, bq, bk, bv, qf, kf, vtf);

  k_attention<<<(BATCH * NHEAD * (SEQ / 16)) / 4, 128, 0, stream>>>(qf, kf, vtf, atf);

  dim3 go(ROWS / 128, D_MODEL / 128);               // (32, 8)
  k_gemm_out<<<go, 256, 0, stream>>>(atf, woh, bo, x, y);
}
